// MotherNet_52604759441783
// MI455X (gfx1250) — compile-verified, matrix operand order checked by reading, not run on hardware
//
#include <hip/hip_runtime.h>

// ---------------------------------------------------------------- types
typedef __bf16 bf16;
typedef __attribute__((ext_vector_type(16))) __bf16 v16bf;
typedef __attribute__((ext_vector_type(8)))  __bf16 v8bf;
typedef __attribute__((ext_vector_type(8)))  float  v8f;

__device__ inline bf16 f2bf(float f) {
    union { float f; unsigned u; } a; a.f = f;
    unsigned r = (a.u + 0x7FFFu + ((a.u >> 16) & 1u)) >> 16;
    union { unsigned short s; bf16 b; } o; o.s = (unsigned short)r;
    return o.b;
}
__device__ inline float bf2f(bf16 b) {
    union { unsigned short s; bf16 b; } i; i.b = b;
    union { unsigned u; float f; } o; o.u = ((unsigned)i.s) << 16;
    return o.f;
}
__device__ inline float wave_sum(float v) {
#pragma unroll
    for (int m = 16; m >= 1; m >>= 1) v += __shfl_xor(v, m, 32);
    return v;
}
__device__ inline float wave_max(float v) {
#pragma unroll
    for (int m = 16; m >= 1; m >>= 1) v = fmaxf(v, __shfl_xor(v, m, 32));
    return v;
}

// ---------------------------------------------------------------- WMMA core
// 64x64 fp32 tile per wave: acc += A[64xK] * Bt[64xK]^T via bf16 WMMA.
// 16 v_wmma per 8 fragment loads per 32-wide K step (~33 FLOP/byte from L2).
__device__ inline void wave_tile_64x64(const bf16* __restrict__ A, int lda,
                                       const bf16* __restrict__ Bt, int ldb,
                                       int K, int lane, v8f (&acc)[4][4]) {
    const int r    = lane & 15;
    const int koff = (lane & 16) ? 8 : 0;
    for (int k = 0; k < K; k += 32) {
        v16bf fa[4], fb[4];
#pragma unroll
        for (int i = 0; i < 4; ++i) {
            const bf16* p = A + (long)(i * 16 + r) * lda + k + koff;
            union { v16bf v; v8bf h[2]; } u;
            u.h[0] = *(const v8bf*)p;
            u.h[1] = *(const v8bf*)(p + 16);
            fa[i] = u.v;
        }
#pragma unroll
        for (int j = 0; j < 4; ++j) {
            const bf16* p = Bt + (long)(j * 16 + r) * ldb + k + koff;
            union { v16bf v; v8bf h[2]; } u;
            u.h[0] = *(const v8bf*)p;
            u.h[1] = *(const v8bf*)(p + 16);
            fb[j] = u.v;
        }
#pragma unroll
        for (int i = 0; i < 4; ++i)
#pragma unroll
            for (int j = 0; j < 4; ++j)
                acc[i][j] = __builtin_amdgcn_wmma_f32_16x16x32_bf16(
                    false, fa[i], false, fb[j], (short)0, acc[i][j], false, false);
    }
}

// act: 0 = none, 1 = exact gelu. C points at tile origin (m0,n0); bias by abs n.
__device__ inline void store_tile_64x64(bf16* __restrict__ C, int ldc, int lane,
                                        const v8f (&acc)[4][4],
                                        const float* __restrict__ bias, int n0,
                                        int act, float scale) {
    const int r  = lane & 15;
    const int hi = (lane >> 4) & 1;
#pragma unroll
    for (int i = 0; i < 4; ++i)
#pragma unroll
        for (int j = 0; j < 4; ++j)
#pragma unroll
            for (int v = 0; v < 8; ++v) {
                int m = i * 16 + v + hi * 8;
                int n = j * 16 + r;
                float x = acc[i][j][v] * scale;
                if (bias) x += bias[n0 + n];
                if (act == 1) x = 0.5f * x * (1.0f + erff(x * 0.70710678118f));
                C[(long)m * ldc + n] = f2bf(x);
            }
}

// ---------------------------------------------------------------- GEMM kernels
// 8 waves arranged waves_m x (8/waves_m); block tile = (64*waves_m) x (64*waves_n).
__global__ __launch_bounds__(256) void k_gemm(const bf16* __restrict__ A, int lda,
                                              const bf16* __restrict__ Bt, int ldb,
                                              bf16* __restrict__ C, int ldc,
                                              const float* __restrict__ bias,
                                              int M, int N, int K, int act,
                                              int waves_m) {
    int wave = threadIdx.x >> 5, lane = threadIdx.x & 31;
    int wm = wave % waves_m, wn = wave / waves_m;
    int m0 = (blockIdx.y * waves_m + wm) * 64;
    int n0 = (blockIdx.x * (8 / waves_m) + wn) * 64;
    if (m0 + 64 > M || n0 + 64 > N) return;
    v8f acc[4][4] = {};
    wave_tile_64x64(A + (long)m0 * lda, lda, Bt + (long)n0 * ldb, ldb, K, lane, acc);
    store_tile_64x64(C + (long)m0 * ldc + n0, ldc, lane, acc, bias, n0, act, 1.0f);
}

// scores[z, t, s] = 0.125 * sum_d q[t,z,d] k[s,z,d]   (z = b*8+h)
__global__ __launch_bounds__(256) void k_attn_scores(const bf16* __restrict__ qkv,
                                                     bf16* __restrict__ attn) {
    int wave = threadIdx.x >> 5, lane = threadIdx.x & 31;
    int z = blockIdx.z, b = z >> 3, hh = z & 7;
    int m0 = (blockIdx.y * 2 + (wave & 1)) * 64;      // 2 m-waves
    int n0 = (blockIdx.x * 4 + (wave >> 1)) * 64;     // 4 n-waves
    const bf16* A  = qkv + b * 1536 + hh * 64;        // q slice
    const bf16* Bt = qkv + b * 1536 + 512 + hh * 64;  // k slice (N x K)
    bf16* C = attn + (long)z * 512 * 512;
    v8f acc[4][4] = {};
    wave_tile_64x64(A + (long)m0 * 24576, 24576, Bt + (long)n0 * 24576, 24576, 64, lane, acc);
    store_tile_64x64(C + (long)m0 * 512 + n0, 512, lane, acc, nullptr, n0, 0, 0.125f);
}

// o[t, b, h*64+d] = sum_s attn[z,t,s] vT[z,d,s]   (8x1 wave layout: 512x64 block)
__global__ __launch_bounds__(256) void k_attn_av(const bf16* __restrict__ attn,
                                                 const bf16* __restrict__ vT,
                                                 bf16* __restrict__ o) {
    int wave = threadIdx.x >> 5, lane = threadIdx.x & 31;
    int z = blockIdx.z, b = z >> 3, hh = z & 7;
    int m0 = wave * 64;                               // all 8 waves over M
    const bf16* A  = attn + (long)z * 512 * 512;
    const bf16* Bt = vT + (long)z * 64 * 512;
    bf16* C = o + b * 512 + hh * 64;                  // ldc = 8192
    v8f acc[4][4] = {};
    wave_tile_64x64(A + (long)m0 * 512, 512, Bt, 512, 512, lane, acc);
    store_tile_64x64(C + (long)m0 * 8192, 8192, lane, acc, nullptr, 0, 0, 1.0f);
}

// ---------------------------------------------------------------- helpers
// fp32 [R x C] -> bf16 [C x R] (batched over z)
__global__ __launch_bounds__(256) void k_transpose(const float* __restrict__ in,
                                                   bf16* __restrict__ out, int R, int C) {
    __shared__ float tile[32][33];
    long zoff = (long)blockIdx.z * R * C;
    in += zoff; out += zoff;
    int c0 = blockIdx.x * 32, r0 = blockIdx.y * 32;
    for (int i = threadIdx.y; i < 32; i += 8) {
        int r = r0 + i, c = c0 + threadIdx.x;
        tile[i][threadIdx.x] = (r < R && c < C) ? in[(long)r * C + c] : 0.0f;
    }
    __syncthreads();
    for (int i = threadIdx.y; i < 32; i += 8) {
        int orow = c0 + i, oc = r0 + threadIdx.x;
        if (orow < C && oc < R) out[(long)orow * R + oc] = f2bf(tile[threadIdx.x][i]);
    }
}

// x_src = x @ enc_W + enc_b ; train part += y-enc -> h (bf16); eval part -> qeval (fp32)
__global__ __launch_bounds__(256) void k_embed(const float* __restrict__ x,
                                               const float* __restrict__ y,
                                               const float* __restrict__ W,
                                               const float* __restrict__ eb,
                                               const float* __restrict__ yw,
                                               const float* __restrict__ yb,
                                               bf16* __restrict__ h,
                                               float* __restrict__ qeval) {
    long idx = (long)blockIdx.x * 256 + threadIdx.x;
    if (idx >= 10240l * 512) return;
    int e = (int)(idx & 511);
    long row = idx >> 9;                      // row = t*16 + b
    const float* xr = x + row * 100;
    float acc = eb[e];
    for (int f = 0; f < 100; ++f) acc += xr[f] * W[f * 512 + e];
    if (row < 8192) {
        h[row * 512 + e] = f2bf(acc + y[row] * yw[e] + yb[e]);
    } else {
        qeval[(row - 8192) * 512 + e] = acc;
    }
}

// vT[z, d, t] = qkv[t*16+b, 1024 + h*64 + d]
__global__ __launch_bounds__(256) void k_vt(const bf16* __restrict__ qkv,
                                            bf16* __restrict__ vT) {
    long idx = (long)blockIdx.x * 256 + threadIdx.x;
    if (idx >= 128l * 64 * 512) return;
    int t = (int)(idx & 511);
    long r = idx >> 9;
    int d = (int)(r & 63);
    int z = (int)(r >> 6), b = z >> 3, hh = z & 7;
    vT[idx] = qkv[(long)(t * 16 + b) * 1536 + 1024 + hh * 64 + d];
}

// wave-per-row softmax over 512, in place, bf16 storage / fp32 math
__global__ __launch_bounds__(256) void k_softmax(bf16* __restrict__ attn) {
    long row = (long)blockIdx.x * 8 + (threadIdx.x >> 5);
    int lane = threadIdx.x & 31;
    bf16* p = attn + row * 512;
    float v[16]; float mx = -3.0e30f;
#pragma unroll
    for (int i = 0; i < 16; ++i) { v[i] = bf2f(p[lane + i * 32]); mx = fmaxf(mx, v[i]); }
    mx = wave_max(mx);
    float s = 0.0f;
#pragma unroll
    for (int i = 0; i < 16; ++i) { v[i] = __expf(v[i] - mx); s += v[i]; }
    s = wave_sum(s);
    float inv = 1.0f / s;
#pragma unroll
    for (int i = 0; i < 16; ++i) p[lane + i * 32] = f2bf(v[i] * inv);
}

// out = LayerNorm(x + a) * g + b   (wave per 512-wide row; in-place safe)
__global__ __launch_bounds__(256) void k_add_ln(const bf16* __restrict__ x,
                                                const bf16* __restrict__ a,
                                                bf16* __restrict__ out,
                                                const float* __restrict__ g,
                                                const float* __restrict__ bv) {
    long row = (long)blockIdx.x * 8 + (threadIdx.x >> 5);
    int lane = threadIdx.x & 31;
    const bf16* xr = x + row * 512;
    const bf16* ar = a + row * 512;
    float v[16]; float s = 0.0f;
#pragma unroll
    for (int i = 0; i < 16; ++i) {
        float t = bf2f(xr[lane + i * 32]) + bf2f(ar[lane + i * 32]);
        v[i] = t; s += t;
    }
    float mean = wave_sum(s) * (1.0f / 512.0f);
    float s2 = 0.0f;
#pragma unroll
    for (int i = 0; i < 16; ++i) { float d = v[i] - mean; s2 += d * d; }
    float rstd = rsqrtf(wave_sum(s2) * (1.0f / 512.0f) + 1e-5f);
#pragma unroll
    for (int i = 0; i < 16; ++i) {
        int e = lane + i * 32;
        out[row * 512 + e] = f2bf((v[i] - mean) * rstd * g[e] + bv[e]);
    }
}

// pooled[b,e] = mean_t h[t,b,e]
__global__ __launch_bounds__(256) void k_pool(const bf16* __restrict__ h,
                                              float* __restrict__ pooled) {
    int idx = blockIdx.x * 256 + threadIdx.x;
    if (idx >= 8192) return;
    int e = idx & 511, b = idx >> 9;
    float s = 0.0f;
    for (int t = 0; t < 512; ++t) s += bf2f(h[(long)t * 8192 + b * 512 + e]);
    pooled[idx] = s * (1.0f / 512.0f);
}

// act1 = relu(pooled @ dec1_W + dec1_b)   (16x512 @ 512x1024)
__global__ __launch_bounds__(256) void k_dec1(const float* __restrict__ pooled,
                                              const float* __restrict__ W,
                                              const float* __restrict__ bias,
                                              float* __restrict__ act1) {
    int idx = blockIdx.x * 256 + threadIdx.x;
    if (idx >= 16 * 1024) return;
    int j = idx & 1023, b = idx >> 10;
    float acc = bias[j];
    for (int k = 0; k < 512; ++k) acc += pooled[b * 512 + k] * W[k * 1024 + j];
    act1[idx] = fmaxf(acc, 0.0f);
}

// flat = act1 @ dec2_W + dec2_b  (16x1024 @ 1024x83466) — weight-streaming fp32
__global__ __launch_bounds__(256) void k_dec2(const float* __restrict__ act,
                                              const float* __restrict__ W,
                                              const float* __restrict__ bias,
                                              float* __restrict__ flat) {
    __shared__ float sa[256 * 16];
    int j = blockIdx.x * 256 + threadIdx.x;
    float acc[16];
#pragma unroll
    for (int b = 0; b < 16; ++b) acc[b] = 0.0f;
    for (int k0 = 0; k0 < 1024; k0 += 256) {
        __syncthreads();
#pragma unroll
        for (int i = 0; i < 16; ++i) {
            int e = threadIdx.x + i * 256;          // e = kk*16 + b
            sa[e] = act[(e & 15) * 1024 + k0 + (e >> 4)];
        }
        __syncthreads();
        if (j < 83466) {
            for (int kk = 0; kk < 256; ++kk) {
                float w = W[(long)(k0 + kk) * 83466 + j];
#pragma unroll
                for (int b = 0; b < 16; ++b) acc[b] += sa[kk * 16 + b] * w;
            }
        }
    }
    if (j < 83466) {
        float bj = bias[j];
#pragma unroll
        for (int b = 0; b < 16; ++b) flat[(long)b * 83466 + j] = acc[b] + bj;
    }
}

// per-batch hyper-net MLP on eval tokens (fp32, tiny)
__global__ __launch_bounds__(256) void k_head1(const float* __restrict__ q,
                                               const float* __restrict__ flat,
                                               float* __restrict__ z1) {
    long idx = (long)blockIdx.x * 256 + threadIdx.x;
    if (idx >= 128l * 16 * 128) return;
    int hh = (int)(idx & 127);
    long r = idx >> 7;
    int b = (int)(r & 15), e = (int)(r >> 4);
    const float* qp = q + ((long)e * 16 + b) * 512;
    const float* fb = flat + (long)b * 83466;
    float acc = fb[65536 + hh];                      // b1
    for (int f = 0; f < 512; ++f) acc += qp[f] * fb[f * 128 + hh];
    z1[idx] = fmaxf(acc, 0.0f);
}
__global__ __launch_bounds__(256) void k_head2(const float* __restrict__ z1,
                                               const float* __restrict__ flat,
                                               float* __restrict__ z2) {
    long idx = (long)blockIdx.x * 256 + threadIdx.x;
    if (idx >= 128l * 16 * 128) return;
    int hh = (int)(idx & 127);
    long r = idx >> 7;
    int b = (int)(r & 15);
    const float* zp = z1 + (r << 7);
    const float* fb = flat + (long)b * 83466;
    float acc = fb[82048 + hh];                      // b2
    for (int f = 0; f < 128; ++f) acc += zp[f] * fb[65664 + f * 128 + hh];
    z2[idx] = fmaxf(acc, 0.0f);
}
__global__ __launch_bounds__(256) void k_head3(const float* __restrict__ z2,
                                               const float* __restrict__ flat,
                                               float* __restrict__ out) {
    long idx = (long)blockIdx.x * 256 + threadIdx.x;
    if (idx >= 128l * 16 * 10) return;
    int o = (int)(idx % 10);
    long r = idx / 10;
    int b = (int)(r & 15);
    const float* zp = z2 + (r << 7);
    const float* fb = flat + (long)b * 83466;
    float acc = fb[83456 + o];                       // bo
    for (int f = 0; f < 128; ++f) acc += zp[f] * fb[82176 + f * 10 + o];
    out[idx] = acc;
}

// ---------------------------------------------------------------- launch
extern "C" void kernel_launch(void* const* d_in, const int* in_sizes, int n_in,
                              void* d_out, int out_size, void* d_ws, size_t ws_size,
                              hipStream_t stream) {
    (void)in_sizes; (void)n_in; (void)out_size; (void)ws_size;
    const float* x_org  = (const float*)d_in[0];
    const float* y_src  = (const float*)d_in[1];
    const float* enc_W  = (const float*)d_in[2];
    const float* enc_b  = (const float*)d_in[3];
    const float* yenc_w = (const float*)d_in[4];
    const float* yenc_b = (const float*)d_in[5];
    const float* qkv_W  = (const float*)d_in[6];
    const float* qkv_b  = (const float*)d_in[7];
    const float* out_W  = (const float*)d_in[8];
    const float* out_b  = (const float*)d_in[9];
    const float* ln1_g  = (const float*)d_in[10];
    const float* ln1_b  = (const float*)d_in[11];
    const float* ln2_g  = (const float*)d_in[12];
    const float* ln2_b  = (const float*)d_in[13];
    const float* ff1_W  = (const float*)d_in[14];
    const float* ff1_b  = (const float*)d_in[15];
    const float* ff2_W  = (const float*)d_in[16];
    const float* ff2_b  = (const float*)d_in[17];
    const float* dec1_W = (const float*)d_in[18];
    const float* dec1_b = (const float*)d_in[19];
    const float* dec2_W = (const float*)d_in[20];
    const float* dec2_b = (const float*)d_in[21];
    float* out = (float*)d_out;

    char* base = (char*)d_ws;
    size_t off = 0;
    auto carve = [&](size_t bytes) -> void* {
        void* p = base + off;
        off += (bytes + 255) & ~(size_t)255;
        return p;
    };
    bf16* qkvWt = (bf16*)carve(12ull * 1536 * 512 * 2);
    bf16* outWt = (bf16*)carve(12ull * 512 * 512 * 2);
    bf16* ff1Wt = (bf16*)carve(12ull * 1024 * 512 * 2);
    bf16* ff2Wt = (bf16*)carve(12ull * 512 * 1024 * 2);
    bf16* h     = (bf16*)carve(8192ull * 512 * 2);
    float* qev  = (float*)carve(2048ull * 512 * 4);
    bf16* qkvb  = (bf16*)carve(8192ull * 1536 * 2);
    bf16* attn  = (bf16*)carve(128ull * 512 * 512 * 2);
    bf16* vT    = (bf16*)carve(128ull * 64 * 512 * 2);
    bf16* obuf  = (bf16*)carve(8192ull * 512 * 2);
    bf16* abuf  = (bf16*)carve(8192ull * 512 * 2);
    bf16* ffb   = (bf16*)carve(8192ull * 1024 * 2);
    float* pool = (float*)carve(8192ull * 4);
    float* act1 = (float*)carve(16384ull * 4);
    float* flat = (float*)carve(16ull * 83466 * 4);
    float* z1   = (float*)carve(262144ull * 4);
    float* z2   = (float*)carve(262144ull * 4);

    dim3 blk(256);
    // weight transposes fp32 -> bf16 (Bt layout, batched over 12 layers)
    k_transpose<<<dim3(48, 16, 12), dim3(32, 8), 0, stream>>>(qkv_W, qkvWt, 512, 1536);
    k_transpose<<<dim3(16, 16, 12), dim3(32, 8), 0, stream>>>(out_W, outWt, 512, 512);
    k_transpose<<<dim3(32, 16, 12), dim3(32, 8), 0, stream>>>(ff1_W, ff1Wt, 512, 1024);
    k_transpose<<<dim3(16, 32, 12), dim3(32, 8), 0, stream>>>(ff2_W, ff2Wt, 1024, 512);

    k_embed<<<20480, blk, 0, stream>>>(x_org, y_src, enc_W, enc_b, yenc_w, yenc_b, h, qev);

    for (int l = 0; l < 12; ++l) {
        const bf16* qWt = qkvWt + (size_t)l * 1536 * 512;
        const bf16* oWt = outWt + (size_t)l * 512 * 512;
        const bf16* f1Wt = ff1Wt + (size_t)l * 1024 * 512;
        const bf16* f2Wt = ff2Wt + (size_t)l * 512 * 1024;

        // blocks: 128 rows (waves_m=2) x 256 cols (4 n-waves), 64x64 per wave
        k_gemm<<<dim3(6, 64, 1), blk, 0, stream>>>(h, 512, qWt, 512, qkvb, 1536,
                                                   qkv_b + l * 1536, 8192, 1536, 512, 0, 2);
        k_vt<<<16384, blk, 0, stream>>>(qkvb, vT);
        k_attn_scores<<<dim3(2, 4, 128), blk, 0, stream>>>(qkvb, attn);
        k_softmax<<<8192, blk, 0, stream>>>(attn);
        k_attn_av<<<dim3(1, 1, 128), blk, 0, stream>>>(attn, vT, obuf);
        k_gemm<<<dim3(2, 64, 1), blk, 0, stream>>>(obuf, 512, oWt, 512, abuf, 512,
                                                   out_b + l * 512, 8192, 512, 512, 0, 2);
        k_add_ln<<<1024, blk, 0, stream>>>(h, abuf, h, ln1_g + l * 512, ln1_b + l * 512);
        k_gemm<<<dim3(4, 64, 1), blk, 0, stream>>>(h, 512, f1Wt, 512, ffb, 1024,
                                                   ff1_b + l * 1024, 8192, 1024, 512, 1, 2);
        k_gemm<<<dim3(2, 64, 1), blk, 0, stream>>>(ffb, 1024, f2Wt, 1024, abuf, 512,
                                                   ff2_b + l * 512, 8192, 512, 1024, 0, 2);
        k_add_ln<<<1024, blk, 0, stream>>>(h, abuf, h, ln2_g + l * 512, ln2_b + l * 512);
    }

    k_pool<<<32, blk, 0, stream>>>(h, pool);
    k_dec1<<<64, blk, 0, stream>>>(pool, dec1_W, dec1_b, act1);
    k_dec2<<<327, blk, 0, stream>>>(act1, dec2_W, dec2_b, flat);
    k_head1<<<1024, blk, 0, stream>>>(qev, flat, z1);
    k_head2<<<1024, blk, 0, stream>>>(z1, flat, z2);
    k_head3<<<80, blk, 0, stream>>>(z2, flat, out);
}
